// ConvNAT_83399674954095
// MI455X (gfx1250) — compile-verified, matrix operand order checked by reading, not run on hardware
//
#include <hip/hip_runtime.h>
#include <hip/hip_bf16.h>
#include <math.h>

// ---------------- problem constants (from reference setup_inputs) -----------
#define B_    2
#define H_    56
#define W_    56
#define CIN   64
#define CMID  128
#define HW    3136            // H*W
#define NS    6272            // B*H*W   (GEMM M dimension)
#define NTILE 392             // NS/16
#define KK    7               // kernel_size
// dilation = 2, hardcoded in nbr1()

typedef _Float16 v16h __attribute__((ext_vector_type(16)));
typedef float    v8f  __attribute__((ext_vector_type(8)));
typedef unsigned int u32x4 __attribute__((ext_vector_type(4)));
typedef int          i32x8 __attribute__((ext_vector_type(8)));
typedef int          i32x4 __attribute__((ext_vector_type(4)));

union Frag16 { v16h h; uint4 u[2]; };

// K index held by (lane-half hi, slot t) for 16-bit A/B WMMA fragments
// (ISA 7.12.2: lanes 0-15 hold K 0..7 & 16..23; lanes 16-31 hold 8..15 & 24..31)
__device__ __forceinline__ int kmap(int t, int hi) {
  return (t & 7) + ((t >> 3) << 4) + (hi << 3);
}

// ---------------- prep: positional-encoding tables --------------------------
__global__ void prep_pe_kernel(float* eh, float* ew) {
  int idx = blockIdx.x * 256 + threadIdx.x;
  if (idx >= 2 * 56 * 32) return;
  int axis = idx / (56 * 32);
  int rem  = idx % (56 * 32);
  int pos  = rem / 32;
  int i    = rem % 32;
  float div = expf(-logf(10000.0f) * (2.0f * (float)i) / 64.0f);
  float ang = (float)pos * div;
  float* dst = axis ? ew : eh;
  dst[pos * 64 + 2 * i]     = sinf(ang);
  dst[pos * 64 + 2 * i + 1] = cosf(ang);
}

// ---------------- prep: conv weights -> f16 B-fragment layout ---------------
__global__ void prep_convw_kernel(const float* __restrict__ cw,
                                  _Float16* __restrict__ out) {
  int idx = blockIdx.x * 256 + threadIdx.x;   // exactly 73728 threads
  int t    = idx & 15;
  int lane = (idx >> 4) & 31;
  int nt   = (idx >> 9) & 7;
  int kc   = (idx >> 12) & 1;
  int tap  = idx >> 13;                        // 0..8
  int N  = lane & 15, hi = lane >> 4;
  int K  = kmap(t, hi);
  int cin = kc * 32 + K;
  int co  = nt * 16 + N;
  out[idx] = (_Float16)cw[(co * CIN + cin) * 9 + tap];
}

// ---------------- prep: q/k/v weights -> f16 B-fragment layout --------------
__global__ void prep_qkvw_kernel(const float* __restrict__ qw,
                                 const float* __restrict__ kw,
                                 const float* __restrict__ vw,
                                 _Float16* __restrict__ out) {
  int idx = blockIdx.x * 256 + threadIdx.x;   // exactly 49152 threads
  int t     = idx & 15;
  int lane  = (idx >> 4) & 31;
  int nt    = (idx >> 9) & 7;
  int chunk = (idx >> 12) & 3;
  int m     = idx >> 14;                       // 0..2
  int N  = lane & 15, hi = lane >> 4;
  int K  = kmap(t, hi);
  int c  = chunk * 32 + K;
  int o  = nt * 16 + N;
  const float* Wm = (m == 0) ? qw : ((m == 1) ? kw : vw);
  out[idx] = (_Float16)Wm[o * CMID + c];
}

// ---------------- stage 1: 3x3 conv as implicit GEMM (WMMA f16) -------------
__global__ void conv_wmma_kernel(const float* __restrict__ x,
                                 const float* __restrict__ cb,
                                 const _Float16* __restrict__ wconv,
                                 _Float16* __restrict__ sfA) {
  __shared__ __align__(16) _Float16 At[512];   // 16(M) x 32(K) tile
  int tile = blockIdx.x;
  int tid  = threadIdx.x;
  int wid  = tid >> 5, lane = tid & 31;
  v8f acc = {0.f, 0.f, 0.f, 0.f, 0.f, 0.f, 0.f, 0.f};

#pragma unroll
  for (int tap = 0; tap < 9; ++tap) {
    int dy = tap / 3 - 1, dx = tap % 3 - 1;
#pragma unroll
    for (int kc = 0; kc < 2; ++kc) {
      __syncthreads();
      // cooperative im2col fill of the A tile in fragment order
      for (int e = tid; e < 512; e += 256) {
        int t = e & 15, ln = e >> 4;
        int M = ln & 15, hi = ln >> 4;
        int cin = kc * 32 + kmap(t, hi);
        int s = tile * 16 + M;
        int b = s / HW; int r2 = s - b * HW;
        int h = r2 / W_, w = r2 - (r2 / W_) * W_;
        int y = h + dy, xx = w + dx;
        float val = 0.0f;
        if (y >= 0 && y < H_ && xx >= 0 && xx < W_)
          val = x[((b * CIN + cin) * H_ + y) * W_ + xx];
        At[e] = (_Float16)val;
      }
      __syncthreads();
      Frag16 a, bf;
      const uint4* ap = (const uint4*)At;
      a.u[0] = ap[lane * 2];
      a.u[1] = ap[lane * 2 + 1];
      const uint4* bp =
          (const uint4*)(wconv + ((((tap * 2 + kc) * 8 + wid) * 32 + lane) << 4));
      bf.u[0] = bp[0];
      bf.u[1] = bp[1];
      acc = __builtin_amdgcn_wmma_f32_16x16x32_f16(
          false, a.h, false, bf.h, (short)0, acc, false, false);
    }
  }

  int hi = lane >> 4, nl = lane & 15;
  int co = wid * 16 + nl;                      // stage-2 K index
  float bias = cb[co];
#pragma unroll
  for (int r = 0; r < 8; ++r) {
    float val = acc[r] + bias;
    int M = r + hi * 8;
    int g = (co & 31) >> 3;
    int lane2 = M + ((g & 1) << 4);
    int t2 = ((g >> 1) << 3) + (co & 7);
    sfA[((tile * 4 + (co >> 5)) * 32 + lane2) * 16 + t2] = (_Float16)val;
  }
}

// ---------------- stage 2: fused Q/K/V GEMMs (WMMA f16) ---------------------
// The block's 4 KB of A-fragments are DMA'd once into LDS by the Tensor Data
// Mover (one tensor_load_to_lds per block), then all 8 waves consume from LDS.
__global__ void qkv_wmma_kernel(const _Float16* __restrict__ sfA,
                                const _Float16* __restrict__ wqkv,
                                const float* __restrict__ qb_,
                                const float* __restrict__ kb_,
                                const float* __restrict__ vb_,
                                float* __restrict__ qo,
                                float* __restrict__ ko,
                                float* __restrict__ vo) {
  __shared__ __align__(16) _Float16 Asm[2048];  // 4 chunks x 32 lanes x 16 f16
  int tile = blockIdx.x;
  int tid = threadIdx.x, wid = tid >> 5, lane = tid & 31;

  if (wid == 0) {
    // D# group 0: count=1, lds_addr, 57-bit global addr, type=2 ("image")
    u32x4 g0;
    unsigned long long ga =
        (unsigned long long)(uintptr_t)sfA + (unsigned long long)tile * 4096ull;
    g0[0] = 1u;                                           // count=1, user mode
    g0[1] = (unsigned int)(uintptr_t)(&Asm[0]);           // LDS byte offset
    g0[2] = (unsigned int)ga;                             // global_addr[31:0]
    g0[3] = (unsigned int)((ga >> 32) & 0x01FFFFFFu) | (2u << 30); // [56:32]|type
    // D# group 1: data_size=8B, 1-D tile of 512 units (4 KB), stride 512
    i32x8 g1;
    g1[0] = 0x00030000;   // wg_mask=0, data_size=3 (8B), no pad/iterate
    g1[1] = 0x02000000;   // tensor_dim0[15:0]=512 in bits[31:16]
    g1[2] = 0x00010000;   // tensor_dim0 hi=0, tensor_dim1 lo16=1
    g1[3] = 0x02000000;   // tensor_dim1 hi=0, tile_dim0=512
    g1[4] = 0x00000001;   // tile_dim1=1, tile_dim2=0
    g1[5] = 512;          // tensor_dim0_stride lo32
    g1[6] = 0;            // stride hi16 / dim1_stride lo16
    g1[7] = 0;            // dim1_stride hi32
    i32x4 z4 = {0, 0, 0, 0};                              // groups 2/3: unused
    i32x8 z8 = {0, 0, 0, 0, 0, 0, 0, 0};
    __builtin_amdgcn_tensor_load_to_lds(g0, g1, z4, z4, z8, 0);
    __builtin_amdgcn_s_wait_tensorcnt(0);
  }
  __syncthreads();

  v8f a0 = {0.f,0.f,0.f,0.f,0.f,0.f,0.f,0.f};
  v8f a1 = a0, a2 = a0;

#pragma unroll
  for (int chunk = 0; chunk < 4; ++chunk) {
    Frag16 a;
    const uint4* ap = (const uint4*)(Asm + ((chunk * 32 + lane) << 4));
    a.u[0] = ap[0];
    a.u[1] = ap[1];
    Frag16 b0, b1, b2;
    const uint4* p0 = (const uint4*)(wqkv + ((((0 * 4 + chunk) * 8 + wid) * 32 + lane) << 4));
    const uint4* p1 = (const uint4*)(wqkv + ((((1 * 4 + chunk) * 8 + wid) * 32 + lane) << 4));
    const uint4* p2 = (const uint4*)(wqkv + ((((2 * 4 + chunk) * 8 + wid) * 32 + lane) << 4));
    b0.u[0] = p0[0]; b0.u[1] = p0[1];
    b1.u[0] = p1[0]; b1.u[1] = p1[1];
    b2.u[0] = p2[0]; b2.u[1] = p2[1];
    a0 = __builtin_amdgcn_wmma_f32_16x16x32_f16(false, a.h, false, b0.h, (short)0, a0, false, false);
    a1 = __builtin_amdgcn_wmma_f32_16x16x32_f16(false, a.h, false, b1.h, (short)0, a1, false, false);
    a2 = __builtin_amdgcn_wmma_f32_16x16x32_f16(false, a.h, false, b2.h, (short)0, a2, false, false);
  }

  int hi = lane >> 4, nl = lane & 15;
  int o = wid * 16 + nl;
  float bq = qb_[o], bk = kb_[o], bv = vb_[o];
#pragma unroll
  for (int r = 0; r < 8; ++r) {
    int s = tile * 16 + r + hi * 8;
    qo[s * CMID + o] = a0[r] + bq;
    ko[s * CMID + o] = a1[r] + bk;
    vo[s * CMID + o] = a2[r] + bv;
  }
}

// ---------------- stage 3: dilated neighborhood attention (VALU f32) --------
__device__ __forceinline__ int nbr1(int i, int j) {
  int p = i >> 1, r = i & 1;
  int st = p - 3;
  st = st < 0 ? 0 : (st > 21 ? 21 : st);
  int idx = r + (st + j) * 2;
  return idx > 55 ? 55 : idx;
}

__device__ __forceinline__ float nbr_logit(const float* __restrict__ k,
                                           const float* __restrict__ eh,
                                           const float* __restrict__ ew,
                                           const float* __restrict__ qloc,
                                           int b, int h, int w, int n, int* kiOut) {
  int u = n / 7, v = n - (n / 7) * 7;
  int kh = nbr1(h, u), kw = nbr1(w, v);
  int ki = b * HW + kh * W_ + kw;
  *kiOut = ki;
  const float* kr = k + ki * CMID;
  __builtin_prefetch(kr, 0, 1);                // global_prefetch_b8
  float d = 0.0f;
  for (int t = 0; t < 128; ++t) d += qloc[t] * kr[t];
  const float* er = eh + kh * 64;
  for (int t = 0; t < 64; ++t) d += qloc[128 + t] * er[t];
  const float* wr = ew + kw * 64;
  for (int t = 0; t < 64; ++t) d += qloc[192 + t] * wr[t];
  return d * 0.0625f;                          // 1/sqrt(256)
}

__global__ void attn_kernel(const float* __restrict__ q,
                            const float* __restrict__ k,
                            const float* __restrict__ v,
                            const float* __restrict__ eh,
                            const float* __restrict__ ew,
                            float* __restrict__ out) {
  __shared__ float lq[8][256];
  __shared__ float la[8][52];
  __shared__ int   li[8][52];
  int tid = threadIdx.x, wid = tid >> 5, lane = tid & 31;
  int s = blockIdx.x * 8 + wid;                // 784*8 == 6272, always valid
  int b = s / HW; int r2 = s - b * HW;
  int h = r2 / W_, w = r2 - (r2 / W_) * W_;

  for (int t = lane; t < 128; t += 32) lq[wid][t] = q[s * CMID + t];
  for (int t = lane; t < 64; t += 32) {
    lq[wid][128 + t] = eh[h * 64 + t];
    lq[wid][192 + t] = ew[w * 64 + t];
  }
  __syncthreads();

  int ki1 = 0, ki2 = 0;
  float l1 = nbr_logit(k, eh, ew, lq[wid], b, h, w, lane, &ki1);
  float l2 = -INFINITY;
  if (lane < 17) l2 = nbr_logit(k, eh, ew, lq[wid], b, h, w, lane + 32, &ki2);

  float m = fmaxf(l1, l2);
#pragma unroll
  for (int off = 16; off > 0; off >>= 1) m = fmaxf(m, __shfl_xor(m, off, 32));
  float e1 = expf(l1 - m);
  float e2 = (lane < 17) ? expf(l2 - m) : 0.0f;
  float ssum = e1 + e2;
#pragma unroll
  for (int off = 16; off > 0; off >>= 1) ssum += __shfl_xor(ssum, off, 32);
  float inv = 1.0f / ssum;

  la[wid][lane] = e1 * inv;
  li[wid][lane] = ki1;
  if (lane < 17) {
    la[wid][lane + 32] = e2 * inv;
    li[wid][lane + 32] = ki2;
  }
  __syncthreads();

  float a0 = 0.f, a1 = 0.f, a2 = 0.f, a3 = 0.f;
  for (int n = 0; n < 49; ++n) {
    float wg = la[wid][n];
    const float* vr = v + li[wid][n] * CMID;
    __builtin_prefetch(vr, 0, 1);
    a0 += wg * vr[lane];
    a1 += wg * vr[lane + 32];
    a2 += wg * vr[lane + 64];
    a3 += wg * vr[lane + 96];
  }
  out[s * CMID + lane]      = a0;
  out[s * CMID + lane + 32] = a1;
  out[s * CMID + lane + 64] = a2;
  out[s * CMID + lane + 96] = a3;
}

// ---------------- launcher ---------------------------------------------------
extern "C" void kernel_launch(void* const* d_in, const int* in_sizes, int n_in,
                              void* d_out, int out_size, void* d_ws, size_t ws_size,
                              hipStream_t stream) {
  const float* x  = (const float*)d_in[0];
  const float* cw = (const float*)d_in[1];
  const float* cb = (const float*)d_in[2];
  const float* qw = (const float*)d_in[3];
  const float* qb = (const float*)d_in[4];
  const float* kw = (const float*)d_in[5];
  const float* kb = (const float*)d_in[6];
  const float* vw = (const float*)d_in[7];
  const float* vb = (const float*)d_in[8];
  // d_in[9]=kernel_size(7), d_in[10]=dilation(2): fixed by setup, hardcoded.

  char* ws = (char*)d_ws;
  _Float16* wconv = (_Float16*)(ws + 0);         // 147456 B
  _Float16* wqkv  = (_Float16*)(ws + 147456);    //  98304 B
  _Float16* sfA   = (_Float16*)(ws + 245760);    // 1605632 B
  float* qbuf = (float*)(ws + 1851392);          // 3211264 B
  float* kbuf = (float*)(ws + 5062656);          // 3211264 B
  float* vbuf = (float*)(ws + 8273920);          // 3211264 B
  float* eh   = (float*)(ws + 11485184);         //   14336 B
  float* ew   = (float*)(ws + 11499520);         //   14336 B
  float* out  = (float*)d_out;

  prep_pe_kernel  <<<14, 256, 0, stream>>>(eh, ew);
  prep_convw_kernel<<<288, 256, 0, stream>>>(cw, wconv);
  prep_qkvw_kernel<<<192, 256, 0, stream>>>(qw, kw, vw, wqkv);
  conv_wmma_kernel<<<NTILE, 256, 0, stream>>>(x, cb, wconv, sfA);
  qkv_wmma_kernel <<<NTILE, 256, 0, stream>>>(sfA, wqkv, qb, kb, vb, qbuf, kbuf, vbuf);
  attn_kernel     <<<784, 256, 0, stream>>>(qbuf, kbuf, vbuf, eh, ew, out);
}
